// MultiHeadCrossAttention3D_21629455302885
// MI455X (gfx1250) — matrix operand order repaired and checked by reading, not compiled, not run on hardware
//
#include <hip/hip_runtime.h>
#include <hip/hip_bf16.h>

// ---- problem constants ----
#define BB 4
#define NN 4096
#define MM 2048
#define QD 512
#define CD 768
#define HH 8
#define DD 64
#define SCALE 0.125f
#define NEGBIG -1.0e9f

typedef __attribute__((ext_vector_type(16))) __bf16 v16bf;
typedef __attribute__((ext_vector_type(8)))  float v8f;

union AFrag { v16bf v; unsigned int u[8]; };
union Facc  { v8f v; float f[8]; };

__device__ __forceinline__ unsigned short f2bf(float f) {
  unsigned int u = __float_as_uint(f);
  u += 0x7FFFu + ((u >> 16) & 1u);   // round-to-nearest-even
  return (unsigned short)(u >> 16);
}
__device__ __forceinline__ unsigned int pack2bf(float lo, float hi) {
  return (unsigned int)f2bf(lo) | ((unsigned int)f2bf(hi) << 16);
}

// async copy of 16 bytes global -> LDS (tracked by ASYNCcnt)
__device__ __forceinline__ void async_g2l_b128(unsigned lds_off, const void* gptr) {
  asm volatile("global_load_async_to_lds_b128 %0, %1, off"
               :: "v"(lds_off), "v"((unsigned long long)(size_t)gptr) : "memory");
}
__device__ __forceinline__ void wait_async0() {
  asm volatile("s_wait_asynccnt 0x0" ::: "memory");
}

// ---------------- f32 -> bf16 conversion ----------------
__global__ void cvt_bf16_kernel(const float* __restrict__ src,
                                unsigned short* __restrict__ dst, int n) {
  int i = blockIdx.x * 256 + threadIdx.x;
  if (i < n) dst[i] = f2bf(src[i]);
}

// ---- weight convert + transpose: W[K,N] f32 -> Wt[N,K] bf16 ----
__global__ void cvt_w_t_kernel(const float* __restrict__ src,
                               unsigned short* __restrict__ dst,
                               int Kdim, int Ndim) {
  int i = blockIdx.x * 256 + threadIdx.x;
  if (i >= Kdim * Ndim) return;
  int k = i / Ndim, n = i - k * Ndim;
  dst[(size_t)n * Kdim + k] = f2bf(src[i]);
}

// ---------------- tiled bf16 GEMM: C[M,N] = A[M,K] @ Bt[N,K]^T ----------------
// OUTMODE: 0 = bf16 row-major; 1 = f32 + bias; 2 = bf16 per-head d-major (V^T)
template <int OUTMODE>
__global__ __launch_bounds__(128) void gemm_bf16_kernel(
    const unsigned short* __restrict__ A, const unsigned short* __restrict__ Bt,
    void* __restrict__ Cout, const float* __restrict__ bias,
    int Mdim, int Ndim, int Kdim)
{
  __shared__ __align__(16) unsigned short As[64][40];  // [m][k]
  __shared__ __align__(16) unsigned short Bs[64][40];  // [n][k]

  const int tid  = threadIdx.x;
  const int wave = tid >> 5;
  const int lane = tid & 31;
  const int g    = lane >> 4;
  const int l16  = lane & 15;
  const int m0   = blockIdx.y * 64;
  const int n0   = blockIdx.x * 64;

  Facc acc[4];
#pragma unroll
  for (int i = 0; i < 4; ++i)
#pragma unroll
    for (int r = 0; r < 8; ++r) acc[i].f[r] = 0.0f;

  for (int kt = 0; kt < Kdim; kt += 32) {
    __syncthreads();
#pragma unroll
    for (int i = 0; i < 2; ++i) {
      int c = tid + i * 128;
      int row = c >> 2, col8 = (c & 3) << 3;
      async_g2l_b128((unsigned)(size_t)&As[row][col8],
                     A + (size_t)(m0 + row) * Kdim + kt + col8);
      async_g2l_b128((unsigned)(size_t)&Bs[row][col8],
                     Bt + (size_t)(n0 + row) * Kdim + kt + col8);
    }
    wait_async0();
    __syncthreads();

    AFrag bf;   // B fragment (32x16): N = l16, K = g*16 + 2v
#pragma unroll
    for (int v = 0; v < 8; ++v)
      bf.u[v] = *(const unsigned int*)&Bs[wave * 16 + l16][g * 16 + 2 * v];

#pragma unroll
    for (int ms = 0; ms < 4; ++ms) {
      AFrag af;  // A fragment (16x32): M = l16
#pragma unroll
      for (int v = 0; v < 8; ++v) {
        int k0 = g * 16 + 2 * v;
        af.u[v] = *(const unsigned int*)&As[ms * 16 + l16][k0];
      }
      acc[ms].v = __builtin_amdgcn_wmma_f32_16x16x32_bf16(
          false, af.v, false, bf.v, (short)0, acc[ms].v, false, false);
    }
  }

  const int col = n0 + wave * 16 + l16;
#pragma unroll
  for (int ms = 0; ms < 4; ++ms) {
#pragma unroll
    for (int r = 0; r < 8; ++r) {
      int row = m0 + ms * 16 + g * 8 + r;
      float val = acc[ms].f[r];
      if (OUTMODE == 1) {
        ((float*)Cout)[(size_t)row * Ndim + col] = val + bias[col];
      } else if (OUTMODE == 0) {
        ((unsigned short*)Cout)[(size_t)row * Ndim + col] = f2bf(val);
      } else {
        int bb = row >> 11, m = row & (MM - 1);
        int hh = col >> 6,  dd = col & (DD - 1);
        ((unsigned short*)Cout)[(((size_t)(bb * HH + hh) * DD + dd) << 11) + m] =
            f2bf(val);
      }
    }
  }
}

// ---------------- flash attention (transposed-S formulation) ----------------
// S^T = K @ Q^T : per lane, one query column -> softmax stats are lane scalars.
// O^T = V^T @ P^T : epilogue writes packed bf16 pairs.
// grid = (N/128, B*H); block = 256 threads = 8 waves; wave owns 16 query rows.
__global__ __launch_bounds__(256) void attn_kernel(
    const unsigned short* __restrict__ Q,    // (B*N, 512) bf16
    const unsigned short* __restrict__ K,    // (B*M, 512) bf16
    const unsigned short* __restrict__ Vt,   // (B*H*64, M) bf16, d-major
    const int* __restrict__ mask,            // (B, N, M) int32
    unsigned short* __restrict__ Out)        // (B*N, 512) bf16
{
  __shared__ __align__(16) unsigned short Ktile[32][72];  // [key][d]
  __shared__ __align__(16) unsigned short Vtile[64][40];  // [d][key]

  const int nb   = blockIdx.x;
  const int bh   = blockIdx.y;
  const int b    = bh >> 3;
  const int h    = bh & 7;
  const int tid  = threadIdx.x;
  const int wave = tid >> 5;
  const int lane = tid & 31;
  const int g    = lane >> 4;
  const int l16  = lane & 15;

  const int qn = nb * 128 + wave * 16 + l16;        // this lane's query row
  const unsigned short* qb = Q + ((size_t)(b * NN) + qn) * QD + h * DD;
  const int* mrowp = mask + ((size_t)b * NN + qn) * MM;

  // Q^T B-fragments (N = query = l16, K = d = c*32 + g*16 + 2v, pairs contiguous)
  AFrag qf[2];
#pragma unroll
  for (int c = 0; c < 2; ++c)
#pragma unroll
    for (int v = 0; v < 8; ++v)
      qf[c].u[v] = *(const unsigned int*)(qb + c * 32 + g * 16 + 2 * v);

  float mrow = -3.0e38f, lrow = 0.0f;
  Facc o[4];
#pragma unroll
  for (int dt = 0; dt < 4; ++dt)
#pragma unroll
    for (int r = 0; r < 8; ++r) o[dt].f[r] = 0.0f;

  for (int kb = 0; kb < MM; kb += 32) {
    // ---- stage K (32x64) and Vt (64x32) tiles in LDS (async, 1 chunk/thread) ----
    __syncthreads();
    {
      int row = tid >> 3, col16 = (tid & 7) << 3;   // K tile
      async_g2l_b128((unsigned)(size_t)&Ktile[row][col16],
                     K + ((size_t)(b * MM) + kb + row) * QD + h * DD + col16);
      int vrow = tid >> 2, col8 = (tid & 3) << 3;   // Vt tile
      async_g2l_b128((unsigned)(size_t)&Vtile[vrow][col8],
                     Vt + ((size_t)bh * DD + vrow) * MM + kb + col8);
    }
    if (kb + 32 < MM)
      __builtin_prefetch(mrowp + kb + 32 + g * 8, 0, 1);
    wait_async0();
    __syncthreads();

    // ---- S^T = K @ Q^T for 32 keys (two 16-key tiles in M dimension) ----
    Facc s[2];
#pragma unroll
    for (int t = 0; t < 2; ++t) {
      v8f sa = {};
#pragma unroll
      for (int c = 0; c < 2; ++c) {
        AFrag kf;  // A frag: M = key = l16, K = d = c*32 + g*16 + 2v
#pragma unroll
        for (int v = 0; v < 8; ++v) {
          int k0 = g * 16 + 2 * v;
          kf.u[v] = *(const unsigned int*)&Ktile[t * 16 + l16][c * 32 + k0];
        }
        sa = __builtin_amdgcn_wmma_f32_16x16x32_bf16(
            false, kf.v, false, qf[c].v, (short)0, sa, false, false);
      }
      s[t].v = sa;
    }

    // ---- mask + scale (keys are consecutive per lane -> b128 loads) ----
#pragma unroll
    for (int t = 0; t < 2; ++t) {
      const int* mp = mrowp + kb + t * 16 + g * 8;
      int4 ma = *(const int4*)mp;
      int4 mb = *(const int4*)(mp + 4);
      s[t].f[0] = (ma.x == 0) ? NEGBIG : s[t].f[0] * SCALE;
      s[t].f[1] = (ma.y == 0) ? NEGBIG : s[t].f[1] * SCALE;
      s[t].f[2] = (ma.z == 0) ? NEGBIG : s[t].f[2] * SCALE;
      s[t].f[3] = (ma.w == 0) ? NEGBIG : s[t].f[3] * SCALE;
      s[t].f[4] = (mb.x == 0) ? NEGBIG : s[t].f[4] * SCALE;
      s[t].f[5] = (mb.y == 0) ? NEGBIG : s[t].f[5] * SCALE;
      s[t].f[6] = (mb.z == 0) ? NEGBIG : s[t].f[6] * SCALE;
      s[t].f[7] = (mb.w == 0) ? NEGBIG : s[t].f[7] * SCALE;
    }

    // ---- online softmax: lane-local reduce + single xor-16 exchange ----
    float lmax = -3.0e38f;
#pragma unroll
    for (int t = 0; t < 2; ++t)
#pragma unroll
      for (int r = 0; r < 8; ++r) lmax = fmaxf(lmax, s[t].f[r]);
    lmax = fmaxf(lmax, __shfl_xor(lmax, 16));
    float mnew = fmaxf(mrow, lmax);
    float corr = __expf(mrow - mnew);
    mrow = mnew;

    float lsum = 0.0f;
#pragma unroll
    for (int t = 0; t < 2; ++t)
#pragma unroll
      for (int r = 0; r < 8; ++r) {
        float p = __expf(s[t].f[r] - mnew);
        s[t].f[r] = p;
        lsum += p;
      }
    lsum += __shfl_xor(lsum, 16);
    lrow = lrow * corr + lsum;

#pragma unroll
    for (int dt = 0; dt < 4; ++dt)
#pragma unroll
      for (int r = 0; r < 8; ++r) o[dt].f[r] *= corr;

    // ---- build P^T B-fragment: pack bf16 pairs, one xor-16 exchange ----
    unsigned int a0[4], a1[4], b0[4], b1[4];
#pragma unroll
    for (int j = 0; j < 4; ++j) {
      a0[j] = pack2bf(s[0].f[2 * j], s[0].f[2 * j + 1]);  // own tile-0 keys
      a1[j] = pack2bf(s[1].f[2 * j], s[1].f[2 * j + 1]);  // own tile-1 keys
    }
#pragma unroll
    for (int j = 0; j < 4; ++j) {
      b0[j] = __shfl_xor((int)a0[j], 16);   // partner half's keys
      b1[j] = __shfl_xor((int)a1[j], 16);
    }
    AFrag pf;  // B frag of P^T: N = query = l16, K = key = g*16 + 2v
#pragma unroll
    for (int j = 0; j < 4; ++j) {
      pf.u[j]     = g ? b1[j] : a0[j];   // keys g*16 + 0..7
      pf.u[4 + j] = g ? a1[j] : b0[j];   // keys g*16 + 8..15
    }

    // ---- O^T += V^T @ P^T (A frag straight from d-major Vtile) ----
#pragma unroll
    for (int dt = 0; dt < 4; ++dt) {
      AFrag vfa;  // A frag: M = d = l16, K = key = g*16 + 2v
#pragma unroll
      for (int v = 0; v < 8; ++v) {
        int k0 = g * 16 + 2 * v;
        vfa.u[v] = *(const unsigned int*)&Vtile[dt * 16 + l16][k0];
      }
      o[dt].v = __builtin_amdgcn_wmma_f32_16x16x32_bf16(
          false, vfa.v, false, pf.v, (short)0, o[dt].v, false, false);
    }
  }

  // ---- epilogue: normalize, pack bf16 pairs, b32 stores ----
  float inv = 1.0f / lrow;
  unsigned short* ob = Out + ((size_t)(b * NN) + qn) * QD + h * DD;
#pragma unroll
  for (int dt = 0; dt < 4; ++dt)
#pragma unroll
    for (int j = 0; j < 4; ++j) {
      unsigned int pk = pack2bf(o[dt].f[2 * j] * inv, o[dt].f[2 * j + 1] * inv);
      *(unsigned int*)(ob + dt * 16 + g * 8 + 2 * j) = pk;
    }
}

// ---------------- launch ----------------
extern "C" void kernel_launch(void* const* d_in, const int* in_sizes, int n_in,
                              void* d_out, int out_size, void* d_ws, size_t ws_size,
                              hipStream_t stream) {
  (void)in_sizes; (void)n_in; (void)out_size; (void)ws_size;
  const float* x    = (const float*)d_in[0];
  const float* ctx  = (const float*)d_in[1];
  const int*   mask = (const int*)d_in[2];
  const float* Wq   = (const float*)d_in[3];
  const float* Wk   = (const float*)d_in[4];
  const float* Wv   = (const float*)d_in[5];
  const float* Wout = (const float*)d_in[6];
  const float* bout = (const float*)d_in[7];

  unsigned short* ws = (unsigned short*)d_ws;
  size_t off = 0;
  unsigned short* xb    = ws + off; off += (size_t)BB * NN * QD;
  unsigned short* ctxb  = ws + off; off += (size_t)BB * MM * CD;
  unsigned short* wqbT  = ws + off; off += (size_t)QD * QD;   // [N,K]
  unsigned short* wkbT  = ws + off; off += (size_t)QD * CD;
  unsigned short* wvbT  = ws + off; off += (size_t)QD * CD;
  unsigned short* wobT  = ws + off; off += (size_t)QD * QD;
  unsigned short* Qbuf  = ws + off; off += (size_t)BB * NN * QD;
  unsigned short* Kbuf  = ws + off; off += (size_t)BB * MM * QD;
  unsigned short* Vt    = ws + off; off += (size_t)BB * MM * QD;
  unsigned short* AOut  = ws + off; off += (size_t)BB * NN * QD;

  cvt_bf16_kernel<<<(BB * NN * QD + 255) / 256, 256, 0, stream>>>(x, xb, BB * NN * QD);
  cvt_bf16_kernel<<<(BB * MM * CD + 255) / 256, 256, 0, stream>>>(ctx, ctxb, BB * MM * CD);
  cvt_w_t_kernel<<<(QD * QD + 255) / 256, 256, 0, stream>>>(Wq,   wqbT, QD, QD);
  cvt_w_t_kernel<<<(CD * QD + 255) / 256, 256, 0, stream>>>(Wk,   wkbT, CD, QD);
  cvt_w_t_kernel<<<(CD * QD + 255) / 256, 256, 0, stream>>>(Wv,   wvbT, CD, QD);
  cvt_w_t_kernel<<<(QD * QD + 255) / 256, 256, 0, stream>>>(Wout, wobT, QD, QD);

  gemm_bf16_kernel<0><<<dim3(QD / 64, (BB * NN) / 64), 128, 0, stream>>>(
      xb, wqbT, Qbuf, nullptr, BB * NN, QD, QD);
  gemm_bf16_kernel<0><<<dim3(QD / 64, (BB * MM) / 64), 128, 0, stream>>>(
      ctxb, wkbT, Kbuf, nullptr, BB * MM, QD, CD);
  gemm_bf16_kernel<2><<<dim3(QD / 64, (BB * MM) / 64), 128, 0, stream>>>(
      ctxb, wvbT, Vt, nullptr, BB * MM, QD, CD);

  attn_kernel<<<dim3(NN / 128, BB * HH), 256, 0, stream>>>(
      Qbuf, Kbuf, Vt, mask, AOut);

  gemm_bf16_kernel<1><<<dim3(QD / 64, (BB * NN) / 64), 128, 0, stream>>>(
      AOut, wobT, d_out, bout, BB * NN, QD, QD);
}
